// InteractionBlock_59648505807201
// MI455X (gfx1250) — matrix operand order repaired and checked by reading, not compiled, not run on hardware
//
#include <hip/hip_runtime.h>
#include <hip/hip_bf16.h>

#define NN 20000
#define NE 320000

typedef __attribute__((ext_vector_type(16))) _Float16 v16h;
typedef __attribute__((ext_vector_type(8)))  _Float16 v8h;
typedef __attribute__((ext_vector_type(8)))  float    v8f;

// ---------------- helpers ----------------
__device__ __forceinline__ void atomicMaxOrd(unsigned int* p, float f) {
  unsigned int u = __float_as_uint(f);
  unsigned int e = (u & 0x80000000u) ? ~u : (u | 0x80000000u);  // order-preserving encode
  atomicMax(p, e);
}
__device__ __forceinline__ float decOrd(unsigned int e) {
  unsigned int v = (e & 0x80000000u) ? (e & 0x7fffffffu) : ~e;
  return __uint_as_float(v);
}

__global__ void fill_f32(float* p, float v, int n) {
  int i = blockIdx.x * blockDim.x + threadIdx.x;
  if (i < n) p[i] = v;
}

// ---------------- WMMA GEMM: Y[M,N] = X[M,K] @ W[K,N] (+bias)(+accum)(+relu) ----------------
// Block = 128 threads = 4 waves -> 64x16 output tile. A (64xKp) and B (16xKp,
// column-major) staged in LDS as f16, K zero-padded to Kp (mult. of 32) so the
// K-loop is branch-free: 3-4 ds_load_b128 + one v_wmma_f32_16x16x32_f16 per chunk.
// Requires K <= 128 and K % 4 == 0 (all GEMMs here: K in {16,32,64,72,128}).
__global__ void __launch_bounds__(128) gemm_wmma_f16(
    const float* __restrict__ X, const float* __restrict__ W,
    const float* __restrict__ bias, float* __restrict__ Y,
    int M, int K, int N, int relu, int accum) {
  __shared__ _Float16 sA[64 * 128];   // [row][k]  row-major, stride Kp
  __shared__ _Float16 sB[16 * 128];   // [col][k]  col-major, stride Kp
  const int Kp = (K + 31) & ~31;
  const int bn = blockIdx.x << 4;
  const int bm = blockIdx.y << 6;
  const int tid = threadIdx.x;

  // --- stage A tile: coalesced float4 reads, zero-pad OOB rows / k>=K ---
  const int qpr = Kp >> 2;  // float4 quads per row
  for (int idx = tid; idx < 64 * qpr; idx += 128) {
    int r = idx / qpr;
    int kq = (idx - r * qpr) << 2;
    int gr = bm + r;
    float4 v = make_float4(0.f, 0.f, 0.f, 0.f);
    if (gr < M && kq < K) v = *(const float4*)(X + (size_t)gr * K + kq);
    _Float16* p = sA + r * Kp + kq;
    p[0] = (_Float16)v.x; p[1] = (_Float16)v.y;
    p[2] = (_Float16)v.z; p[3] = (_Float16)v.w;
  }
  // --- stage B panel (16 columns), transposed to col-major ---
  for (int idx = tid; idx < 16 * Kp; idx += 128) {
    int k = idx >> 4, c = idx & 15;
    float v = (k < K && bn + c < N) ? W[(size_t)k * N + bn + c] : 0.f;
    sB[c * Kp + k] = (_Float16)v;
  }
  __syncthreads();

  const int lane = tid & 31;
  const int half = lane >> 4, l15 = lane & 15;
  const int wrow = ((tid >> 5) << 4) + l15;                 // row within 64-tile
  const _Float16* aRow = sA + wrow * Kp;
  const _Float16* bCol = sB + l15 * Kp;

  v8f acc = {};
  for (int kk = 0; kk < Kp; kk += 32) {
    // A fragment: elems 0..15 = K [kk+16h, +16) contiguous (matches B)
    v8h a0 = *(const v8h*)(aRow + kk + 16 * half);
    v8h a1 = *(const v8h*)(aRow + kk + 16 * half + 8);
    v16h a = __builtin_shufflevector(a0, a1, 0, 1, 2, 3, 4, 5, 6, 7,
                                     8, 9, 10, 11, 12, 13, 14, 15);
    // B fragment: lane = column l15, elems 0..15 = K [kk+16h, +16) contiguous
    v16h b = *(const v16h*)(bCol + kk + 16 * half);
    acc = __builtin_amdgcn_wmma_f32_16x16x32_f16(false, a, false, b, (short)0, acc,
                                                 false, false);
  }

  const int grow = bm + wrow - l15;  // wave's tile base
  const int bcol = bn + l15;
  const bool bcol_ok = bcol < N;
  float bb = (bias != nullptr && bcol_ok) ? bias[bcol] : 0.f;
#pragma unroll
  for (int j = 0; j < 8; ++j) {
    int row = grow + j + 8 * half;  // C/D layout: VGPR j -> row j (lanes 0-15), j+8 (16-31)
    if (row < M && bcol_ok) {
      size_t o = (size_t)row * N + bcol;
      float v = acc[j] + bb;
      if (accum) v += Y[o];
      if (relu) v = fmaxf(v, 0.f);
      Y[o] = v;
    }
  }
}

// ---------------- GINE: agg[dst] += relu(x[src] + ea@We + be), We staged in LDS ----------------
__global__ void __launch_bounds__(128) gine_scatter_kernel(
    const float* __restrict__ x, const float* __restrict__ ea,
    const int* __restrict__ src, const int* __restrict__ dst,
    const float* __restrict__ We /*32x128*/, const float* __restrict__ be,
    float* __restrict__ agg) {
  __shared__ float sW[32 * 128];
  __shared__ float sb[128];
  __shared__ float sea[32];
  for (int i = threadIdx.x; i < 32 * 128; i += 128) sW[i] = We[i];
  sb[threadIdx.x] = be[threadIdx.x];
  __syncthreads();
  const int EPB = 16;
  int f = threadIdx.x;
  int e0 = blockIdx.x * EPB;
  int e1 = e0 + EPB; if (e1 > NE) e1 = NE;
  for (int e = e0; e < e1; ++e) {
    if (f < 32) sea[f] = ea[(size_t)e * 32 + f];
    __syncthreads();
    int s = src[e], d = dst[e];
    float acc = sb[f];
#pragma unroll 8
    for (int j = 0; j < 32; ++j) acc += sea[j] * sW[j * 128 + f];
    float m = fmaxf(x[(size_t)s * 128 + f] + acc, 0.f);
    atomicAdd(&agg[(size_t)d * 128 + f], m);
    __syncthreads();
  }
}

__global__ void gine_combine(const float* __restrict__ x, const float* __restrict__ eps,
                             float* __restrict__ hid, int n) {
  int i = blockIdx.x * blockDim.x + threadIdx.x;
  if (i < n) hid[i] = (1.f + eps[0]) * x[i] + hid[i];  // hid holds agg; in-place
}

// ---------------- TransformerConv edge passes (H=4, D=16) ----------------
__global__ void tconv_logits(const float* __restrict__ q, const float* __restrict__ kx,
                             const float* __restrict__ ek, const int* __restrict__ src,
                             const int* __restrict__ dst, float* __restrict__ logits,
                             unsigned int* __restrict__ maxb) {
  int t = blockIdx.x * blockDim.x + threadIdx.x;
  if (t >= NE * 4) return;
  int e = t >> 2, h = t & 3;
  int s = src[e], d = dst[e];
  const float* qp = q + (size_t)d * 64 + h * 16;
  const float* kp = kx + (size_t)s * 64 + h * 16;
  const float* ep = ek + (size_t)e * 64 + h * 16;
  float acc = 0.f;
#pragma unroll
  for (int i = 0; i < 16; ++i) acc += qp[i] * (kp[i] + ep[i]);
  acc *= 0.25f;  // 1/sqrt(16)
  logits[t] = acc;
  atomicMaxOrd(&maxb[d * 4 + h], acc);
}

__global__ void seg_exp_sum(const float* __restrict__ logits, const int* __restrict__ dst,
                            const unsigned int* __restrict__ maxb, float* __restrict__ ex,
                            float* __restrict__ sumb, int H) {
  int t = blockIdx.x * blockDim.x + threadIdx.x;
  if (t >= NE * H) return;
  int e = t / H, h = t - e * H;
  int d = dst[e];
  float m = decOrd(maxb[d * H + h]);
  if (!(m > -__builtin_inff() && m < __builtin_inff())) m = 0.f;  // empty seg / NaN -> 0
  float v = __expf(logits[t] - m);
  ex[t] = v;
  atomicAdd(&sumb[d * H + h], v);
}

__global__ void tconv_scatter(const float* __restrict__ ex, const float* __restrict__ sumb,
                              const float* __restrict__ vx, const float* __restrict__ ek,
                              const int* __restrict__ src, const int* __restrict__ dst,
                              float* __restrict__ out) {
  int t = blockIdx.x * blockDim.x + threadIdx.x;
  if (t >= NE * 4) return;
  int e = t >> 2, h = t & 3;
  int s = src[e], d = dst[e];
  float alpha = ex[t] / (sumb[d * 4 + h] + 1e-16f);
  const float* vp = vx + (size_t)s * 64 + h * 16;
  const float* ep = ek + (size_t)e * 64 + h * 16;
  float* op = out + (size_t)d * 64 + h * 16;
#pragma unroll
  for (int i = 0; i < 16; ++i) atomicAdd(&op[i], alpha * (vp[i] + ep[i]));
}

// ---------------- GATv2 edge passes (H=1, D=16) ----------------
__global__ void gat_logits(const float* __restrict__ xl, const float* __restrict__ xr,
                           const float* __restrict__ eW, const float* __restrict__ avec,
                           const int* __restrict__ src, const int* __restrict__ dst,
                           float* __restrict__ elog, unsigned int* __restrict__ maxb) {
  int e = blockIdx.x * blockDim.x + threadIdx.x;
  if (e >= NE) return;
  int s = src[e], d = dst[e];
  float acc = 0.f;
#pragma unroll
  for (int i = 0; i < 16; ++i) {
    float t = xl[(size_t)s * 16 + i] + xr[(size_t)d * 16 + i] + eW[(size_t)e * 16 + i];
    t = (t > 0.f) ? t : 0.2f * t;  // leaky_relu 0.2
    acc += t * avec[i];
  }
  elog[e] = acc;
  atomicMaxOrd(&maxb[d], acc);
}

__global__ void gat_scatter(const float* __restrict__ ex, const float* __restrict__ sumb,
                            const float* __restrict__ xl, const int* __restrict__ src,
                            const int* __restrict__ dst, float* __restrict__ out) {
  int e = blockIdx.x * blockDim.x + threadIdx.x;
  if (e >= NE) return;
  int s = src[e], d = dst[e];
  float alpha = ex[e] / (sumb[d] + 1e-16f);
#pragma unroll
  for (int i = 0; i < 16; ++i)
    atomicAdd(&out[(size_t)d * 16 + i], alpha * xl[(size_t)s * 16 + i]);
}

__global__ void bcast_bias(const float* __restrict__ bias, float* __restrict__ out,
                           int n, int dim) {
  int i = blockIdx.x * blockDim.x + threadIdx.x;
  if (i < n * dim) out[i] = bias[i % dim];
}

// ---------------- final mix ----------------
__global__ void mix_out(const float* __restrict__ x1, const float* __restrict__ x2,
                        const float* __restrict__ x3, const float* __restrict__ w,
                        float* __restrict__ xo, int n) {
  int i = blockIdx.x * blockDim.x + threadIdx.x;
  if (i < n) xo[i] = (x1[i] * w[0] + x2[i] * w[1] + x3[i] * w[2]) * (1.f / 3.f);
}
__global__ void scale_ea(const float* __restrict__ ea, const float* __restrict__ w,
                         float* __restrict__ eo, int n) {
  int i = blockIdx.x * blockDim.x + threadIdx.x;
  if (i < n) eo[i] = ea[i] * ((w[0] + w[1] + w[2]) * (1.f / 3.f));
}

// ---------------- host orchestration ----------------
extern "C" void kernel_launch(void* const* d_in, const int* in_sizes, int n_in,
                              void* d_out, int out_size, void* d_ws, size_t ws_size,
                              hipStream_t stream) {
  const float* x    = (const float*)d_in[0];
  const int*   ei   = (const int*)d_in[1];
  const float* ea   = (const float*)d_in[2];
  const float* mixw = (const float*)d_in[3];
  const int* src = ei;
  const int* dst = ei + NE;
  if (n_in < 4 + 3 * 34) return;  // params flattened in setup_inputs() dict order

  // per-branch leaf order (insertion order of the reference dicts):
  // 0 gine_We, 1 gine_be, 2 gine_eps, 3 mlp_W1, 4 mlp_b1, 5 mlp_W2, 6 mlp_b2,
  // 7..16  t1: Wq,bq,Wk,bk,Wv,bv,We,be,Wskip,bskip
  // 17..26 t2: same
  // 27..33 gat: Wl,bl,Wr,br,We,a,bias
  auto P = [&](int b, int i) -> const float* { return (const float*)d_in[4 + 34 * b + i]; };

  // workspace arena (floats); peak ~124 MB
  float* ws = (float*)d_ws;
  size_t o = 0;
  float* hid128 = ws + o; o += (size_t)NN * 128;   // GINE agg/hid; later aliased: q|kx
  float* q      = hid128;
  float* kx     = hid128 + (size_t)NN * 64;
  float* vx     = ws + o; o += (size_t)NN * 64;    // gat aliases: xl|xr here
  float* xl     = vx;
  float* xr     = vx + (size_t)NN * 16;
  float* hidmid = ws + o; o += (size_t)NN * 72;
  float* h16    = ws + o; o += (size_t)NN * 16;
  float* outA   = ws + o; o += (size_t)NN * 64;
  float* outB   = ws + o; o += (size_t)NN * 64;
  float* ek     = ws + o; o += (size_t)NE * 64;    // gat aliases: eW16 (E*16) | elog (E)
  float* eW16   = ek;
  float* elog   = ek + (size_t)NE * 16;
  float* logits = ws + o; o += (size_t)NE * 4;
  unsigned int* maxb = (unsigned int*)(ws + o); o += (size_t)NN * 4;
  float* sumb   = ws + o; o += (size_t)NN * 4;
  float* xb0    = ws + o; o += (size_t)NN * 16;
  float* xb1    = ws + o; o += (size_t)NN * 16;
  float* xb2    = ws + o; o += (size_t)NN * 16;
  if (ws_size < o * sizeof(float)) return;
  float* xb[3] = {xb0, xb1, xb2};

  auto cdiv = [](int a, int b) { return (a + b - 1) / b; };
  auto gemm = [&](const float* X, const float* W, const float* b, float* Y,
                  int M, int K, int N, int relu, int accum) {
    dim3 g(cdiv(N, 16), cdiv(M, 64));
    gemm_wmma_f16<<<g, dim3(128), 0, stream>>>(X, W, b, Y, M, K, N, relu, accum);
  };
  auto fill = [&](float* p, float v, int n) {
    fill_f32<<<cdiv(n, 256), 256, 0, stream>>>(p, v, n);
  };

  auto tconv = [&](const float* xin, int Kd, int b, int pb, float* out) {
    gemm(xin, P(b, pb + 0), P(b, pb + 1), q,   NN, Kd, 64, 0, 0);  // Wq,bq
    gemm(xin, P(b, pb + 2), P(b, pb + 3), kx,  NN, Kd, 64, 0, 0);  // Wk,bk
    gemm(xin, P(b, pb + 4), P(b, pb + 5), vx,  NN, Kd, 64, 0, 0);  // Wv,bv
    gemm(ea,  P(b, pb + 6), P(b, pb + 7), ek,  NE, 32, 64, 0, 0);  // We,be
    gemm(xin, P(b, pb + 8), P(b, pb + 9), out, NN, Kd, 64, 0, 0);  // Wskip seeds out
    fill((float*)maxb, 0.f, NN * 4);  // encoded-0 == "-inf"/empty
    fill(sumb, 0.f, NN * 4);
    tconv_logits<<<cdiv(NE * 4, 256), 256, 0, stream>>>(q, kx, ek, src, dst, logits, maxb);
    seg_exp_sum<<<cdiv(NE * 4, 256), 256, 0, stream>>>(logits, dst, maxb, logits, sumb, 4);
    tconv_scatter<<<cdiv(NE * 4, 256), 256, 0, stream>>>(logits, sumb, vx, ek, src, dst, out);
  };

  for (int b = 0; b < 3; ++b) {
    // GINE (edge-linear fused, We in LDS) + MLP
    fill(hid128, 0.f, NN * 128);
    gine_scatter_kernel<<<cdiv(NE, 16), 128, 0, stream>>>(x, ea, src, dst,
                                                          P(b, 0), P(b, 1), hid128);
    gine_combine<<<cdiv(NN * 128, 256), 256, 0, stream>>>(x, P(b, 2), hid128, NN * 128);
    gemm(hid128, P(b, 3), P(b, 4), hidmid, NN, 128, 72, 1, 0);
    gemm(hidmid, P(b, 5), P(b, 6), h16,    NN, 72,  16, 0, 0);
    // two TransformerConv layers
    tconv(h16,  16, b, 7,  outA);
    tconv(outA, 64, b, 17, outB);
    // GATv2
    gemm(outB, P(b, 27), P(b, 28), xl,   NN, 64, 16, 0, 0);  // Wl,bl
    gemm(outB, P(b, 29), P(b, 30), xr,   NN, 64, 16, 0, 0);  // Wr,br
    gemm(ea,   P(b, 31), nullptr,  eW16, NE, 32, 16, 0, 0);  // We
    bcast_bias<<<cdiv(NN * 16, 256), 256, 0, stream>>>(P(b, 33), xb[b], NN, 16);
    fill((float*)maxb, 0.f, NN);
    fill(sumb, 0.f, NN);
    gat_logits<<<cdiv(NE, 256), 256, 0, stream>>>(xl, xr, eW16, P(b, 32), src, dst,
                                                  elog, maxb);
    seg_exp_sum<<<cdiv(NE, 256), 256, 0, stream>>>(elog, dst, maxb, elog, sumb, 1);
    gat_scatter<<<cdiv(NE, 256), 256, 0, stream>>>(elog, sumb, xl, src, dst, xb[b]);
  }

  float* xo = (float*)d_out;            // [NN,16]
  float* eo = xo + (size_t)NN * 16;     // [NE,32]
  mix_out<<<cdiv(NN * 16, 256), 256, 0, stream>>>(xb[0], xb[1], xb[2], mixw, xo, NN * 16);
  scale_ea<<<cdiv(NE * 32, 256), 256, 0, stream>>>(ea, mixw, eo, NE * 32);
  (void)in_sizes; (void)out_size;
}
